// AutoEncoderLayer_49366354100290
// MI455X (gfx1250) — compile-verified, matrix operand order checked by reading, not run on hardware
//
#include <hip/hip_runtime.h>
#include <math.h>
#include <stdint.h>

typedef __attribute__((ext_vector_type(16))) __bf16 v16bf;
typedef __attribute__((ext_vector_type(8)))  float  v8f;

typedef unsigned short u16;

#if defined(__gfx1250__)
#define ASYNC_LDS 1
#endif

__device__ __forceinline__ u16 f2bf(float f) {
    unsigned int u = __float_as_uint(f);
    unsigned int lsb = (u >> 16) & 1u;
    u += 0x7fffu + lsb;                 // round-to-nearest-even
    return (u16)(u >> 16);
}

// 16-byte global -> LDS copy: CDNA5 GLOBAL_LOAD_ASYNC_TO_LDS_B128 (ASYNCcnt)
// via inline asm; VGPR round-trip fallback otherwise.
__device__ __forceinline__ void copy16_g2l(const u16* __restrict__ g, u16* l) {
#if defined(ASYNC_LDS)
    unsigned int lds_off = (unsigned int)(uintptr_t)l;  // addr[31:0] = LDS offset
    asm volatile("global_load_async_to_lds_b128 %0, %1, off"
                 :
                 : "v"(lds_off), "v"((const void*)g)
                 : "memory");
#else
    *(uint4*)l = *(const uint4*)g;
#endif
}

__device__ __forceinline__ void wait_async6() {
#if defined(ASYNC_LDS)
    asm volatile("s_wait_asynccnt 0x6" ::: "memory");
#endif
}
__device__ __forceinline__ void wait_async0() {
#if defined(ASYNC_LDS)
    asm volatile("s_wait_asynccnt 0x0" ::: "memory");
#endif
}

// ---------------------------------------------------------------------------
// fp32 -> bf16 convert (grid-stride)
// ---------------------------------------------------------------------------
__global__ void k_cvt_bf16(const float* __restrict__ src, u16* __restrict__ dst, size_t n) {
    size_t i = (size_t)blockIdx.x * blockDim.x + threadIdx.x;
    size_t stride = (size_t)gridDim.x * blockDim.x;
    for (; i < n; i += stride) dst[i] = f2bf(src[i]);
}

// ---------------------------------------------------------------------------
// bf16 transpose: src[R,C] -> dst[C,R]   (32x32 LDS tiles, R,C multiples of 32)
// ---------------------------------------------------------------------------
__global__ __launch_bounds__(256) void k_transpose_bf16(const u16* __restrict__ src,
                                                        u16* __restrict__ dst,
                                                        int R, int C) {
    __shared__ u16 t[32][33];
    const int tx = threadIdx.x & 31;
    const int ty = threadIdx.x >> 5;     // 8 rows of threads
    const int bx = blockIdx.x, by = blockIdx.y;
#pragma unroll
    for (int i = 0; i < 4; ++i) {
        int rr = by * 32 + ty + i * 8;
        t[ty + i * 8][tx] = src[(size_t)rr * C + bx * 32 + tx];
    }
    __syncthreads();
#pragma unroll
    for (int i = 0; i < 4; ++i) {
        int rr = bx * 32 + ty + i * 8;
        dst[(size_t)rr * R + by * 32 + tx] = t[tx][ty + i * 8];
    }
}

// ---------------------------------------------------------------------------
// row softmax: fp32 in[row, n] -> bf16 out[row, n]   (one block per row)
// ---------------------------------------------------------------------------
__global__ __launch_bounds__(256) void k_softmax_bf16(const float* __restrict__ in,
                                                      u16* __restrict__ out, int n) {
    const int row = blockIdx.x;
    const float* p = in + (size_t)row * n;
    __shared__ float red[256];
    const int tid = threadIdx.x;

    float mx = -INFINITY;
    for (int c = tid; c < n; c += 256) mx = fmaxf(mx, p[c]);
    red[tid] = mx; __syncthreads();
    for (int s = 128; s > 0; s >>= 1) {
        if (tid < s) red[tid] = fmaxf(red[tid], red[tid + s]);
        __syncthreads();
    }
    mx = red[0]; __syncthreads();

    float sum = 0.f;
    for (int c = tid; c < n; c += 256) sum += __expf(p[c] - mx);
    red[tid] = sum; __syncthreads();
    for (int s = 128; s > 0; s >>= 1) {
        if (tid < s) red[tid] += red[tid + s];
        __syncthreads();
    }
    const float inv = 1.f / red[0];

    for (int c = tid; c < n; c += 256)
        out[(size_t)row * n + c] = f2bf(__expf(p[c] - mx) * inv);
}

// ---------------------------------------------------------------------------
// LayerNorm + residual: out = LN(in)*g + b + res ; optional bf16 copy of out
// one block per row, n = E
// ---------------------------------------------------------------------------
__global__ __launch_bounds__(256) void k_ln_res(const float* __restrict__ in,
                                                const float* __restrict__ res,
                                                const float* __restrict__ g,
                                                const float* __restrict__ b,
                                                float* __restrict__ out32,
                                                u16* __restrict__ outbf, int n) {
    const int row = blockIdx.x;
    const int tid = threadIdx.x;
    const float* p = in  + (size_t)row * n;
    const float* q = res + (size_t)row * n;
    __shared__ float s1[256], s2[256];

    float sum = 0.f, sq = 0.f;
    for (int c = tid; c < n; c += 256) { float v = p[c]; sum += v; sq += v * v; }
    s1[tid] = sum; s2[tid] = sq; __syncthreads();
    for (int s = 128; s > 0; s >>= 1) {
        if (tid < s) { s1[tid] += s1[tid + s]; s2[tid] += s2[tid + s]; }
        __syncthreads();
    }
    const float mean = s1[0] / n;
    const float var  = s2[0] / n - mean * mean;
    const float rstd = rsqrtf(var + 1e-5f);

    for (int c = tid; c < n; c += 256) {
        float v = (p[c] - mean) * rstd * g[c] + b[c] + q[c];
        out32[(size_t)row * n + c] = v;
        if (outbf) outbf[(size_t)row * n + c] = f2bf(v);
    }
}

// ---------------------------------------------------------------------------
// bf16 WMMA GEMM ("TN"): C[M,N] = scale * A[M,K] * B[N,K]^T (+ bias[N])
// A, B row-major K-contiguous bf16. Output to fp32 C32 or bf16 Cbf.
// Block tile 128x256 (8 waves, wave tile 64x64 = 4x4 WMMA 16x16x32).
// Double-buffered LDS, async global->LDS staging (ASYNCcnt).
// Requires: M%128==0, N%256==0, K%32==0.
// ---------------------------------------------------------------------------
#define BM 128
#define BN 256
#define BK 32

union Frag { uint4 u[2]; v16bf v; };

__global__ __launch_bounds__(256, 1) void k_gemm_bf16(const u16* __restrict__ A,
                                                      const u16* __restrict__ B,
                                                      const float* __restrict__ bias,
                                                      float* __restrict__ C32,
                                                      u16* __restrict__ Cbf,
                                                      int M, int N, int K,
                                                      int lda, int ldb, int ldc,
                                                      float scale) {
    __shared__ __align__(16) u16 As[2][BM * BK];   // 2 x  8 KB
    __shared__ __align__(16) u16 Bs[2][BN * BK];   // 2 x 16 KB

    const int tid   = threadIdx.x;
    const int m0    = blockIdx.y * BM;
    const int n0    = blockIdx.x * BN;
    const int wave  = tid >> 5;
    const int lane  = tid & 31;
    const int half  = lane >> 4;       // which 16-lane half
    const int r     = lane & 15;       // row/col within 16
    const int waveM = wave >> 2;       // 0..1 : 64-row strip
    const int waveN = wave & 3;        // 0..3 : 64-col strip

    v8f acc[4][4];
    const v8f vzero = {};
#pragma unroll
    for (int i = 0; i < 4; ++i)
#pragma unroll
        for (int j = 0; j < 4; ++j) acc[i][j] = vzero;

    // stage one BK-deep tile pair into LDS buffer `buf` (6 x B128 per wave)
    auto stage = [&](int buf, int kk) {
#pragma unroll
        for (int it = 0; it < (BM * BK) / (8 * 256); ++it) {   // 2 chunks
            int c = tid + it * 256;
            int row = c >> 2, col = (c & 3) * 8;
            copy16_g2l(&A[(size_t)(m0 + row) * lda + kk + col], &As[buf][c * 8]);
        }
#pragma unroll
        for (int it = 0; it < (BN * BK) / (8 * 256); ++it) {   // 4 chunks
            int c = tid + it * 256;
            int row = c >> 2, col = (c & 3) * 8;
            copy16_g2l(&B[(size_t)(n0 + row) * ldb + kk + col], &Bs[buf][c * 8]);
        }
    };

    const int nk = K / BK;
    stage(0, 0);

    for (int t = 0; t < nk; ++t) {
        const int cur = t & 1;
        if (t + 1 < nk) {
            stage(cur ^ 1, (t + 1) * BK);
            wait_async6();             // current tile's 6 copies complete
        } else {
            wait_async0();
        }
        __syncthreads();

        // A fragments per ISA 16-bit 16x32 layout:
        // lane half=0 holds K in [0,8)+[16,24), half=1 holds [8,16)+[24,32)
        Frag afr[4];
#pragma unroll
        for (int i = 0; i < 4; ++i) {
            const u16* p = &As[cur][(waveM * 64 + i * 16 + r) * BK + half * 8];
            afr[i].u[0] = *(const uint4*)p;
            afr[i].u[1] = *(const uint4*)(p + 16);
        }
#pragma unroll
        for (int j = 0; j < 4; ++j) {
            Frag bfr;
            const u16* p = &Bs[cur][(waveN * 64 + j * 16 + r) * BK + half * 8];
            bfr.u[0] = *(const uint4*)p;
            bfr.u[1] = *(const uint4*)(p + 16);
#pragma unroll
            for (int i = 0; i < 4; ++i) {
                acc[i][j] = __builtin_amdgcn_wmma_f32_16x16x32_bf16(
                    false, afr[i].v, false, bfr.v, (short)0, acc[i][j], false, false);
            }
        }
        __syncthreads();               // protects buffer reuse two steps later
    }

    // epilogue: C/D layout -> VGPR v: row = v + half*8, col = r (per 16x16 tile)
#pragma unroll
    for (int i = 0; i < 4; ++i) {
#pragma unroll
        for (int j = 0; j < 4; ++j) {
            const int col = n0 + waveN * 64 + j * 16 + r;
            const float bv = bias ? bias[col] : 0.f;
#pragma unroll
            for (int v = 0; v < 8; ++v) {
                const int rowg = m0 + waveM * 64 + i * 16 + half * 8 + v;
                const float val = acc[i][j][v] * scale + bv;
                if (C32) C32[(size_t)rowg * ldc + col] = val;
                else     Cbf[(size_t)rowg * ldc + col] = f2bf(val);
            }
        }
    }
}

// ---------------------------------------------------------------------------
// host orchestration
// ---------------------------------------------------------------------------
extern "C" void kernel_launch(void* const* d_in, const int* in_sizes, int n_in,
                              void* d_out, int out_size, void* d_ws, size_t ws_size,
                              hipStream_t stream) {
    (void)in_sizes; (void)n_in; (void)out_size; (void)ws_size;
    const int S = 4096, E = 1024, H = 8, HE = H * E;

    const float* x  = (const float*)d_in[0];
    const float* Wq = (const float*)d_in[1];
    const float* bq = (const float*)d_in[2];
    const float* Wk = (const float*)d_in[3];
    const float* bk = (const float*)d_in[4];
    const float* Wv = (const float*)d_in[5];
    const float* bv = (const float*)d_in[6];
    const float* Wz = (const float*)d_in[7];
    const float* bz = (const float*)d_in[8];
    const float* g1 = (const float*)d_in[9];
    const float* b1 = (const float*)d_in[10];
    const float* Wf = (const float*)d_in[11];
    const float* bf = (const float*)d_in[12];
    const float* g2 = (const float*)d_in[13];
    const float* b2 = (const float*)d_in[14];
    float* outp = (float*)d_out;

    // workspace carve
    char* w = (char*)d_ws;
    auto carve = [&](size_t bytes) -> void* {
        void* p = (void*)w;
        w += (bytes + 255) & ~(size_t)255;
        return p;
    };
    u16*   xb    = (u16*)  carve((size_t)S * E * 2);
    u16*   Wqb   = (u16*)  carve((size_t)H * E * E * 2);
    u16*   Wkb   = (u16*)  carve((size_t)H * E * E * 2);
    u16*   Wvb   = (u16*)  carve((size_t)H * E * E * 2);
    u16*   Wzb   = (u16*)  carve((size_t)E * HE * 2);
    u16*   Wfb   = (u16*)  carve((size_t)E * E * 2);
    u16*   Zb    = (u16*)  carve((size_t)S * HE * 2);      // concat of heads (bf16)
    float* O32   = (float*)carve((size_t)S * E * 4);
    float* LN1   = (float*)carve((size_t)S * E * 4);
    u16*   LN1b  = (u16*)  carve((size_t)S * E * 2);
    float* FN32  = (float*)carve((size_t)S * E * 4);
    // per-head scratch (reused each head)
    u16*   Qb    = (u16*)  carve((size_t)S * E * 2);
    u16*   Kb    = (u16*)  carve((size_t)S * E * 2);
    u16*   Vb    = (u16*)  carve((size_t)S * E * 2);
    u16*   Qt    = (u16*)  carve((size_t)E * S * 2);
    u16*   Kt    = (u16*)  carve((size_t)E * S * 2);
    float* Sc32  = (float*)carve((size_t)E * E * 4);
    u16*   Ab    = (u16*)  carve((size_t)E * E * 2);
    u16*   At    = (u16*)  carve((size_t)E * E * 2);

    auto cvt = [&](const float* s, u16* d, size_t n) {
        int blocks = (int)((n + 1023) / 1024);
        if (blocks > 4096) blocks = 4096;
        k_cvt_bf16<<<blocks, 256, 0, stream>>>(s, d, n);
    };
    auto gemm = [&](const u16* A, const u16* B, const float* bias,
                    float* C32, u16* Cbf, int M, int N, int K,
                    int lda, int ldb, int ldc, float scale) {
        dim3 grid(N / BN, M / BM);
        k_gemm_bf16<<<grid, 256, 0, stream>>>(A, B, bias, C32, Cbf,
                                              M, N, K, lda, ldb, ldc, scale);
    };
    auto transp = [&](const u16* s, u16* d, int R, int C) {
        dim3 grid(C / 32, R / 32);
        k_transpose_bf16<<<grid, 256, 0, stream>>>(s, d, R, C);
    };

    // 1) operand conversions to bf16
    cvt(x,  xb,  (size_t)S * E);
    cvt(Wq, Wqb, (size_t)H * E * E);
    cvt(Wk, Wkb, (size_t)H * E * E);
    cvt(Wv, Wvb, (size_t)H * E * E);
    cvt(Wz, Wzb, (size_t)E * HE);
    cvt(Wf, Wfb, (size_t)E * E);

    const float inv_sqrtE = 0.03125f; // 1/sqrt(1024)

    // 2) per-head attention
    for (int h = 0; h < H; ++h) {
        const u16* Wqh = Wqb + (size_t)h * E * E;
        const u16* Wkh = Wkb + (size_t)h * E * E;
        const u16* Wvh = Wvb + (size_t)h * E * E;

        // Q/K/V = x @ W_h^T + b_h    -> bf16 [S,E]
        gemm(xb, Wqh, bq + h * E, nullptr, Qb, S, E, E, E, E, E, 1.f);
        gemm(xb, Wkh, bk + h * E, nullptr, Kb, S, E, E, E, E, E, 1.f);
        gemm(xb, Wvh, bv + h * E, nullptr, Vb, S, E, E, E, E, E, 1.f);

        // transposes for the K^T @ Q contraction over S
        transp(Qb, Qt, S, E);   // Qt[f,s] = Q[s,f]
        transp(Kb, Kt, S, E);   // Kt[e,s] = K[s,e]

        // scores[e,f] = (1/sqrt(E)) * sum_s K[s,e] Q[s,f]  (M=E, N=E, K=S)
        gemm(Kt, Qt, nullptr, Sc32, nullptr, E, E, S, S, S, E, inv_sqrtE);

        // softmax over f, emit bf16; then transpose for use as B operand
        k_softmax_bf16<<<E, 256, 0, stream>>>(Sc32, Ab, E);
        transp(Ab, At, E, E);   // At[f,e] = A[e,f]

        // Zh[s,f] = sum_e V[s,e] A[e,f] -> write into concat slot (ldc = H*E)
        gemm(Vb, At, nullptr, nullptr, Zb + (size_t)h * E, S, E, E, E, E, HE, 1.f);
    }

    // 3) O = Z @ Wz^T + bz   (K = 8192)
    gemm(Zb, Wzb, bz, O32, nullptr, S, E, HE, HE, HE, E, 1.f);

    // 4) LN1 = LN(O)*g1+b1 + x  (fp32 + bf16 copy)
    k_ln_res<<<S, 256, 0, stream>>>(O32, x, g1, b1, LN1, LN1b, E);

    // 5) FN = LN1 @ Wf^T + bf
    gemm(LN1b, Wfb, bf, FN32, nullptr, S, E, E, E, E, E, 1.f);

    // 6) LN2 = LN(FN)*g2+b2 + LN1  -> output
    k_ln_res<<<S, 256, 0, stream>>>(FN32, LN1, g2, b2, outp, nullptr, E);
}
